// VAE_RotatedMNIST_71502615544597
// MI455X (gfx1250) — matrix operand order repaired and checked by reading, not compiled, not run on hardware
//
#include <hip/hip_runtime.h>

typedef __attribute__((ext_vector_type(16))) _Float16 v16h;
typedef __attribute__((ext_vector_type(8)))  _Float16 v8h;
typedef __attribute__((ext_vector_type(4)))  _Float16 v4h;
typedef __attribute__((ext_vector_type(8)))  float    v8f;
typedef __attribute__((ext_vector_type(4)))  float    v4f;
typedef __attribute__((ext_vector_type(4)))  int      v4i;

constexpr int E_CNT  = 5;
constexpr int B_ROWS = 65536;
constexpr int IN_D   = 784;
constexpr int HID_N  = 400;
constexpr int LAT_N  = 128;
constexpr int TILE_M = 128;     // rows per block (halves per-block W traffic vs 64)
constexpr int KT     = 32;      // WMMA K per step (f16)
constexpr int NCHUNK = 256;     // columns per pass (16 n-tiles)
constexpr int MPAD   = B_ROWS + E_CNT * TILE_M;   // 66176, segments 128-aligned

// ------------------------------------------------ CDNA5 async global->LDS copy
#if defined(__has_builtin)
#if __has_builtin(__builtin_amdgcn_global_load_async_to_lds_b128)
#define ASYNC_LDS 1
#endif
#endif
#ifndef ASYNC_LDS
#define ASYNC_LDS 0
#endif

__device__ __forceinline__ void stage16_to_lds(const _Float16* gsrc, _Float16* lds) {
#if ASYNC_LDS
    __builtin_amdgcn_global_load_async_to_lds_b128((v4i*)gsrc, (v4i*)lds, 0, 0);
#else
    *(v8h*)lds = *(const v8h*)gsrc;
#endif
}

__device__ __forceinline__ void stage_wait() {
#if ASYNC_LDS
#if __has_builtin(__builtin_amdgcn_s_wait_asynccnt)
    __builtin_amdgcn_s_wait_asynccnt(0);
#else
    asm volatile("s_wait_asynccnt 0x0" ::: "memory");
#endif
#endif
}

// ---------------------------------------------------------------- routing prep

__global__ void init_ctrl_perm(int* ctrl, int* perm, int mpad) {
    int i = blockIdx.x * blockDim.x + threadIdx.x;
    if (i < 32)   ctrl[i] = 0;
    if (i < mpad) perm[i] = -1;
}

__global__ void hist_kernel(const int* __restrict__ idx, int* __restrict__ counts, int n) {
    int i = blockIdx.x * blockDim.x + threadIdx.x;
    if (i < n) atomicAdd(&counts[idx[i]], 1);
}

__global__ void prefix_kernel(const int* __restrict__ counts, int* __restrict__ poff) {
    if (blockIdx.x == 0 && threadIdx.x == 0) {
        int off = 0;
        for (int e = 0; e < E_CNT; ++e) {
            poff[e] = off;
            off += (counts[e] + TILE_M - 1) & ~(TILE_M - 1);
        }
        poff[E_CNT] = off;
    }
}

__global__ void assign_kernel(const int* __restrict__ idx, const int* __restrict__ poff,
                              int* __restrict__ cursors, int* __restrict__ perm, int n) {
    int i = blockIdx.x * blockDim.x + threadIdx.x;
    if (i < n) {
        int e = idx[i];
        int s = atomicAdd(&cursors[e], 1);
        perm[poff[e] + s] = i;
    }
}

// gather x rows into permuted f16 buffer; padding slots get zeros
__global__ void gather_kernel(const float* __restrict__ x, const int* __restrict__ perm,
                              _Float16* __restrict__ xg) {
    int slot = blockIdx.x;
    int row  = perm[slot];
    int c4   = threadIdx.x;                 // 196 groups of 4
    if (c4 < IN_D / 4) {
        v4f v = {0.f, 0.f, 0.f, 0.f};
        if (row >= 0) v = ((const v4f*)(x + (long)row * IN_D))[c4];
        v4h h = { (_Float16)v.x, (_Float16)v.y, (_Float16)v.z, (_Float16)v.w };
        ((v4h*)(xg + (long)slot * IN_D))[c4] = h;
    }
}

// ---------------------------------------------------------------- WMMA GEMM
// Block = 256 threads = 8 waves; wave w owns rows [16w, 16w+16) of a 128-row tile
// and all n-tiles of the current 256-column chunk (guards are block-uniform).
// EPI: 0 = relu -> f16   1 = identity -> f16   2 = identity -> f32
//      3 = sigmoid+clip -> f32, scattered to original rows via perm
template <int EPI, int K, int N>
__global__ __launch_bounds__(256)
void gemm_routed(const _Float16* __restrict__ A,
                 const float* __restrict__ W, const float* __restrict__ bias,
                 long wstride, long bstride,
                 const int* __restrict__ poff,
                 void* __restrict__ out, float* __restrict__ scatter_out,
                 const int* __restrict__ perm) {
    const int m0 = blockIdx.x * TILE_M;
    if (m0 >= poff[E_CNT]) return;                 // uniform exit past padded total
    int e = 0;
    while (e < E_CNT - 1 && m0 >= poff[e + 1]) ++e;
    const float* We = W + (long)e * wstride;
    const float* be = bias + (long)e * bstride;

    __shared__ _Float16 sA[TILE_M * KT];           // [row][k]   8 KB
    __shared__ _Float16 sW[NCHUNK * KT];           // [n][k]    16 KB (transposed)

    const int tid  = threadIdx.x;
    const int lane = tid & 31;
    // wave-uniform values forced into SGPRs
    const int wv   = __builtin_amdgcn_readfirstlane(tid >> 5);
    const int msub = wv * 16;                      // 8 row-groups of 16
    const int lrow = lane & 15;
    const int hi16 = lane >> 4;                    // half-wave select

    int orow[8];
    if (EPI == 3) {
#pragma unroll
        for (int v = 0; v < 8; ++v) orow[v] = perm[m0 + msub + hi16 * 8 + v];
    }

    const v8f vzero = {0.f, 0.f, 0.f, 0.f, 0.f, 0.f, 0.f, 0.f};
    const int sa_r  = tid >> 1;                    // A staging: row per thread
    const int sa_k  = (tid & 1) * 16;              // 16 halfs (32 B) per thread
    const int off1  = hi16 * 16;                   // A-frag K base (contiguous K)
    const int kb    = hi16 * 16;                   // B-frag K base (contiguous K)

#pragma unroll 1
    for (int nc = 0; nc < N; nc += NCHUNK) {
        v8f acc[16];
#pragma unroll
        for (int i = 0; i < 16; ++i) acc[i] = vzero;

#pragma unroll 1
        for (int kt = 0; kt < K; kt += KT) {
            // ---- stage A tile (async DMA to LDS when available)
            {
                const _Float16* src = A + (long)(m0 + sa_r) * K + kt + sa_k;
                _Float16* dst = &sA[sa_r * KT + sa_k];
                if (kt + sa_k + 16 <= K) {         // full 32 B granule in range
                    stage16_to_lds(src, dst);
                    stage16_to_lds(src + 8, dst + 8);
                } else {
                    v8h a0, a1;
#pragma unroll
                    for (int j = 0; j < 8; ++j) {
                        a0[j] = (kt + sa_k + j < K) ? src[j] : (_Float16)0.0f;
                        a1[j] = (kt + sa_k + 8 + j < K) ? src[8 + j] : (_Float16)0.0f;
                    }
                    *(v8h*)dst = a0;
                    *(v8h*)(dst + 8) = a1;
                }
            }
            // ---- stage W tile transposed: b128 loads along N, f16 stores to [n][k]
            {
                int flat = tid;
#pragma unroll
                for (int it = 0; it < 8; ++it, flat += 256) {
                    int kk = flat >> 6;            // 0..31
                    int n4 = (flat & 63) << 2;     // 0..252
                    int kg = kt + kk;
                    int ng = nc + n4;
                    v4f w4 = {0.f, 0.f, 0.f, 0.f};
                    if (kg < K && ng < N)
                        w4 = *(const v4f*)&We[(long)kg * N + ng];
                    sW[(n4 + 0) * KT + kk] = (_Float16)w4.x;
                    sW[(n4 + 1) * KT + kk] = (_Float16)w4.y;
                    sW[(n4 + 2) * KT + kk] = (_Float16)w4.z;
                    sW[(n4 + 3) * KT + kk] = (_Float16)w4.w;
                }
            }
            stage_wait();
            __syncthreads();

            // ---- A fragment (lanes<16: K 0-15; lanes>=16: K 16-31)
            const _Float16* arow = &sA[(msub + lrow) * KT];
            v8h alo = *(const v8h*)(arow + off1);
            v8h ahi = *(const v8h*)(arow + 8 + off1);
            v16h afrag = __builtin_shufflevector(alo, ahi,
                0, 1, 2, 3, 4, 5, 6, 7, 8, 9, 10, 11, 12, 13, 14, 15);

            // ---- 4 quads: batch-load 4 B fragments, then 4 WMMAs back-to-back
#pragma unroll
            for (int q = 0; q < 4; ++q) {
                v16h bf[4];
#pragma unroll
                for (int j = 0; j < 4; ++j) {
                    int nt = q * 4 + j;            // block-uniform guard
                    if (nc + nt * 16 < N)
                        bf[j] = *(const v16h*)&sW[(nt * 16 + lrow) * KT + kb];
                }
#pragma unroll
                for (int j = 0; j < 4; ++j) {
                    int nt = q * 4 + j;
                    if (nc + nt * 16 < N)
                        acc[nt] = __builtin_amdgcn_wmma_f32_16x16x32_f16(
                            false, afrag, false, bf[j], (short)0, acc[nt], false, false);
                }
            }
            __syncthreads();
        }

        // ---- epilogue: acc[i][v] -> row m0+msub+v+8*hi16, col nc+i*16+lrow
#pragma unroll
        for (int i = 0; i < 16; ++i) {
            int n_g = nc + i * 16 + lrow;
            if (nc + i * 16 >= N) continue;        // block-uniform (N % 16 == 0)
            float bv = be[n_g];
#pragma unroll
            for (int v = 0; v < 8; ++v) {
                long row = m0 + msub + hi16 * 8 + v;
                float val = acc[i][v] + bv;
                if (EPI == 0) {
                    val = val > 0.f ? val : 0.f;
                    ((_Float16*)out)[row * N + n_g] = (_Float16)val;
                } else if (EPI == 1) {
                    ((_Float16*)out)[row * N + n_g] = (_Float16)val;
                } else if (EPI == 2) {
                    ((float*)out)[row * N + n_g] = val;
                } else {
                    int r = orow[v];
                    if (r >= 0) {
                        float s = 1.f / (1.f + __expf(-val));
                        s = fminf(fmaxf(s, 0.f), 1.f);
                        scatter_out[(long)r * N + n_g] = s;
                    }
                }
            }
        }
    }
}

// ---------------------------------------------------------------- reparam
__global__ void reparam_kernel(const float* __restrict__ h2, const float* __restrict__ eps,
                               const int* __restrict__ perm,
                               float* __restrict__ out_mu, float* __restrict__ out_lv,
                               _Float16* __restrict__ zf, int mpad) {
    int i = blockIdx.x * blockDim.x + threadIdx.x;
    if (i >= mpad * LAT_N) return;
    int slot = i / LAT_N, j = i - slot * LAT_N;
    float mu = h2[(long)slot * (2 * LAT_N) + j];
    float lv = h2[(long)slot * (2 * LAT_N) + LAT_N + j];
    int row = perm[slot];
    float ev = 0.0f;
    if (row >= 0) {
        ev = eps[(long)row * LAT_N + j];
        out_mu[(long)row * LAT_N + j] = mu;
        out_lv[(long)row * LAT_N + j] = lv;
    }
    zf[(long)slot * LAT_N + j] = (_Float16)(mu + ev * __expf(0.5f * lv));
}

// ---------------------------------------------------------------- launch
extern "C" void kernel_launch(void* const* d_in, const int* in_sizes, int n_in,
                              void* d_out, int out_size, void* d_ws, size_t ws_size,
                              hipStream_t stream) {
    (void)in_sizes; (void)n_in; (void)out_size; (void)ws_size;
    const float* x       = (const float*)d_in[0];
    const int*   aidx    = (const int*)d_in[1];
    const float* eps     = (const float*)d_in[2];
    const float* enc_W1  = (const float*)d_in[3];
    const float* enc_b1  = (const float*)d_in[4];
    const float* enc_W2  = (const float*)d_in[5];
    const float* enc_b2  = (const float*)d_in[6];
    const float* align_W = (const float*)d_in[7];
    const float* align_b = (const float*)d_in[8];
    const float* dec_W1  = (const float*)d_in[9];
    const float* dec_b1  = (const float*)d_in[10];
    const float* dec_W2  = (const float*)d_in[11];
    const float* dec_b2  = (const float*)d_in[12];

    float* out_mu = (float*)d_out;
    float* out_lv = out_mu + (size_t)B_ROWS * LAT_N;
    float* out_xr = out_lv + (size_t)B_ROWS * LAT_N;

    uint8_t* ws = (uint8_t*)d_ws;
    size_t off = 0;
    auto carve = [&](size_t bytes) -> void* {
        void* p = ws + off;
        off = (off + bytes + 255) & ~(size_t)255;
        return p;
    };
    int*      ctrl = (int*)carve(64 * sizeof(int));            // counts / cursors / poff
    int*      perm = (int*)carve((size_t)MPAD * sizeof(int));
    _Float16* xg   = (_Float16*)carve((size_t)MPAD * IN_D * sizeof(_Float16));  // 104 MB
    _Float16* h1   = (_Float16*)carve((size_t)MPAD * HID_N * sizeof(_Float16)); // 53 MB
    _Float16* zf   = (_Float16*)carve((size_t)MPAD * LAT_N * sizeof(_Float16)); // 17 MB
    // dead-interval aliases inside region A / region B:
    float*    h2 = (float*)xg;       // written after xg fully consumed (enc1)
    _Float16* d1 = (_Float16*)xg;    // written after h2 fully consumed (reparam)
    _Float16* za = h1;               // written after h1 fully consumed (enc2)

    int* counts  = ctrl;
    int* cursors = ctrl + 8;
    int* poff    = ctrl + 16;

    init_ctrl_perm<<<(MPAD + 255) / 256, 256, 0, stream>>>(ctrl, perm, MPAD);
    hist_kernel<<<(B_ROWS + 255) / 256, 256, 0, stream>>>(aidx, counts, B_ROWS);
    prefix_kernel<<<1, 32, 0, stream>>>(counts, poff);
    assign_kernel<<<(B_ROWS + 255) / 256, 256, 0, stream>>>(aidx, poff, cursors, perm, B_ROWS);
    gather_kernel<<<MPAD, 256, 0, stream>>>(x, perm, xg);

    const int gtiles = MPAD / TILE_M;
    // enc1: [Mpad x 784] @ [784 x 400] + b, relu -> h1 (f16)
    gemm_routed<0, IN_D, HID_N><<<gtiles, 256, 0, stream>>>(xg, enc_W1, enc_b1,
        (long)IN_D * HID_N, (long)HID_N, poff, h1, nullptr, perm);
    // enc2: [Mpad x 400] @ [400 x 256] + b -> h2 (f32)
    gemm_routed<2, HID_N, 2 * LAT_N><<<gtiles, 256, 0, stream>>>(h1, enc_W2, enc_b2,
        (long)HID_N * 2 * LAT_N, (long)(2 * LAT_N), poff, h2, nullptr, perm);
    // reparam: scatter mu/logvar, z = mu + eps*exp(0.5*lv) -> zf (f16, permuted)
    reparam_kernel<<<(MPAD * LAT_N + 255) / 256, 256, 0, stream>>>(
        h2, eps, perm, out_mu, out_lv, zf, MPAD);
    // align (unrouted: stride 0): [Mpad x 128] @ [128 x 128] + b -> za (f16)
    gemm_routed<1, LAT_N, LAT_N><<<gtiles, 256, 0, stream>>>(zf, align_W, align_b,
        0L, 0L, poff, za, nullptr, perm);
    // dec1: [Mpad x 128] @ [128 x 400] + b, relu -> d1 (f16)
    gemm_routed<0, LAT_N, HID_N><<<gtiles, 256, 0, stream>>>(za, dec_W1, dec_b1,
        (long)LAT_N * HID_N, (long)HID_N, poff, d1, nullptr, perm);
    // dec2: [Mpad x 400] @ [400 x 784] + b, sigmoid+clip, scatter -> x_recon (f32)
    gemm_routed<3, HID_N, IN_D><<<gtiles, 256, 0, stream>>>(d1, dec_W2, dec_b2,
        (long)HID_N * IN_D, (long)IN_D, poff, nullptr, out_xr, perm);
}
